// AGFLayer_14096082665752
// MI455X (gfx1250) — matrix operand order repaired and checked by reading, not compiled, not run on hardware
//
#include <hip/hip_runtime.h>
#include <hip/hip_bf16.h>

typedef __bf16 bf16;
typedef __attribute__((ext_vector_type(16))) __bf16 v16bf;
typedef __attribute__((ext_vector_type(8)))  float  v8f;

struct alignas(16) B128 { unsigned int w[4]; };
union Frag16 { v16bf v; B128 q[2]; };

#define NB 4
#define NN 2048
#define DIMM 512
#define HH 8
#define HD 64
#define BH (NB*HH)        // 32
#define MTOK (NB*NN)      // 8192

__device__ __forceinline__ unsigned short f2bf(float f) {
    unsigned int u = __float_as_uint(f);
    u += 0x7fffu + ((u >> 16) & 1u);
    return (unsigned short)(u >> 16);
}
__device__ __forceinline__ float bf2f(unsigned short s) {
    return __uint_as_float(((unsigned int)s) << 16);
}

// A-fragment (16x32 bf16): lane holds row, elems 0..7 = K kb..kb+7, 8..15 = K kb+16..kb+23, kb = 8*(lane>=16)
__device__ __forceinline__ void load_a16(Frag16& f, const bf16* base, int row, int ld, int lane) {
    int kb = (lane >> 4) << 3;
    const bf16* p = base + row * ld + kb;
    f.q[0] = *reinterpret_cast<const B128*>(p);
    f.q[1] = *reinterpret_cast<const B128*>(p + 16);
}
// B-fragment (32x16 bf16): lane holds column, elems 0..7 = K kb..kb+7, 8..15 = K kb+16..kb+23, kb = 8*(lane>=16)
__device__ __forceinline__ void load_b16(Frag16& f, const bf16* base, int col, int ld, int lane) {
    int kb = (lane >> 4) << 3;
    const bf16* p = base + col * ld + kb;
    f.q[0] = *reinterpret_cast<const B128*>(p);
    f.q[1] = *reinterpret_cast<const B128*>(p + 16);
}

__device__ __forceinline__ v8f wmma_bf(const Frag16& a, const Frag16& b, v8f c) {
    return __builtin_amdgcn_wmma_f32_16x16x32_bf16(false, a.v, false, b.v, (short)0, c, false, false);
}

__device__ __forceinline__ v8f vzero8() {
    v8f z = {0.f,0.f,0.f,0.f,0.f,0.f,0.f,0.f};
    return z;
}

// ---------------- conversion kernels ----------------

__global__ void agf_cvt_x(const float* __restrict__ x, unsigned short* __restrict__ xb, int n) {
    int i = blockIdx.x * blockDim.x + threadIdx.x;
    if (i < n) xb[i] = f2bf(x[i]);
}

__global__ void agf_cvt_w(const float* __restrict__ Wq, const float* __restrict__ Wk,
                          const float* __restrict__ Wv, const float* __restrict__ Wo,
                          unsigned short* __restrict__ WTall) {
    int i = blockIdx.x * blockDim.x + threadIdx.x;   // 512*512 flat (o-major)
    int z = blockIdx.y;
    const float* W = (z == 0) ? Wq : (z == 1) ? Wk : (z == 2) ? Wv : Wo;
    int o = i >> 9, d = i & 511;
    WTall[z * (DIMM * DIMM) + i] = f2bf(W[d * DIMM + o]);   // WT[o][d] = W[d][o]
}

// ---------------- projection GEMM: Q,K,V ----------------
// 16x64 tile per wave: A fragment reused across 4 column accumulators.
// grid (DIMM/64=8 col groups, MTOK/16=512 row tiles, 3), block 32
__global__ void agf_gemm_proj(const unsigned short* __restrict__ xb,
                              const unsigned short* __restrict__ WTall,
                              const float* __restrict__ bq, const float* __restrict__ bk,
                              const float* __restrict__ bv,
                              unsigned short* __restrict__ Qh, unsigned short* __restrict__ Kh,
                              unsigned short* __restrict__ VT) {
    int lane = threadIdx.x & 31;
    int col0 = blockIdx.x * 64;
    int row0 = blockIdx.y * 16;
    int z = blockIdx.z;
    const bf16* A  = (const bf16*)xb;
    const bf16* Bw = (const bf16*)(WTall + z * (DIMM * DIMM));
    const float* bias = (z == 0) ? bq : (z == 1) ? bk : bv;

    int rowa = row0 + (lane & 15);
    v8f acc[4];
#pragma unroll
    for (int j = 0; j < 4; ++j) acc[j] = vzero8();

    for (int kk = 0; kk < DIMM; kk += 32) {
        Frag16 fa;
        load_a16(fa, A + kk, rowa, DIMM, lane);
#pragma unroll
        for (int j = 0; j < 4; ++j) {
            Frag16 fb;
            load_b16(fb, Bw + kk, col0 + j * 16 + (lane & 15), DIMM, lane);
            acc[j] = wmma_bf(fa, fb, acc[j]);
        }
    }
    float scale = (z == 0) ? 0.125f : 1.0f;   // fold 1/sqrt(HD) into Q
#pragma unroll
    for (int j = 0; j < 4; ++j) {
        int col = col0 + j * 16 + (lane & 15);
        float bval = bias[col];
        int h = col >> 6, hd = col & 63;
#pragma unroll
        for (int r = 0; r < 8; ++r) {
            float vl = (acc[j][r] + bval) * scale;
            int token = row0 + r + ((lane >> 4) << 3);
            int bidx = token >> 11, nn = token & 2047;
            int bh = bidx * HH + h;
            unsigned short us = f2bf(vl);
            if (z == 2)       VT[(bh * HD + hd) * NN + nn] = us;      // [b,h,hd,n]
            else if (z == 0)  Qh[(bh * NN + nn) * HD + hd] = us;      // [b,h,n,hd]
            else              Kh[(bh * NN + nn) * HD + hd] = us;
        }
    }
}

// ---------------- softmax stats (online max/sum) ----------------
// 32-row tile per wave: K-tile fragments shared across both row halves.
// grid (NN/32=64 row tiles, 32 bh), block 32
__global__ void agf_attn_stats(const unsigned short* __restrict__ Qh,
                               const unsigned short* __restrict__ Kh,
                               float* __restrict__ mbuf, float* __restrict__ lbuf) {
    int lane = threadIdx.x & 31;
    int R0 = blockIdx.x * 32;
    int bh = blockIdx.y;
    const bf16* qb = (const bf16*)(Qh + bh * NN * HD);
    const bf16* kb = (const bf16*)(Kh + bh * NN * HD);

    Frag16 aq[2][2];
#pragma unroll
    for (int hf = 0; hf < 2; ++hf) {
        int rowa = R0 + hf * 16 + (lane & 15);
        load_a16(aq[hf][0], qb,      rowa, HD, lane);
        load_a16(aq[hf][1], qb + 32, rowa, HD, lane);
    }

    float mr[2][8], lr[2][8];
#pragma unroll
    for (int hf = 0; hf < 2; ++hf)
#pragma unroll
        for (int r = 0; r < 8; ++r) { mr[hf][r] = -3.0e38f; lr[hf][r] = 0.f; }

    for (int ct = 0; ct < NN / 16; ++ct) {
        int key = ct * 16 + (lane & 15);
        Frag16 b0, b1;
        load_b16(b0, kb,      key, HD, lane);
        load_b16(b1, kb + 32, key, HD, lane);
#pragma unroll
        for (int hf = 0; hf < 2; ++hf) {
            v8f lg = vzero8();
            lg = wmma_bf(aq[hf][0], b0, lg);
            lg = wmma_bf(aq[hf][1], b1, lg);
#pragma unroll
            for (int r = 0; r < 8; ++r) {
                float x = lg[r];
                float tm = x;
                tm = fmaxf(tm, __shfl_xor(tm, 1, 32));
                tm = fmaxf(tm, __shfl_xor(tm, 2, 32));
                tm = fmaxf(tm, __shfl_xor(tm, 4, 32));
                tm = fmaxf(tm, __shfl_xor(tm, 8, 32));
                float nm = fmaxf(mr[hf][r], tm);
                float p = __expf(x - nm);
                p += __shfl_xor(p, 1, 32);
                p += __shfl_xor(p, 2, 32);
                p += __shfl_xor(p, 4, 32);
                p += __shfl_xor(p, 8, 32);
                lr[hf][r] = lr[hf][r] * __expf(mr[hf][r] - nm) + p;
                mr[hf][r] = nm;
            }
        }
    }
    if ((lane & 15) == 0) {
#pragma unroll
        for (int hf = 0; hf < 2; ++hf)
#pragma unroll
            for (int r = 0; r < 8; ++r) {
                int row = R0 + hf * 16 + r + ((lane >> 4) << 3);
                mbuf[bh * NN + row] = mr[hf][r];
                lbuf[bh * NN + row] = lr[hf][r];
            }
    }
}

// out = c0 * V (fp32 accumulator init), layout [b,h,hd,n]
__global__ void agf_init_out(const unsigned short* __restrict__ VT,
                             float* __restrict__ outacc, const float* __restrict__ coeffs) {
    int i = blockIdx.x * blockDim.x + threadIdx.x;   // BH*HD*NN = 4,194,304
    int bh = i >> 17;                                // HD*NN = 131072
    int h = bh & 7;
    outacc[i] = coeffs[h * 4] * bf2f(VT[i]);
}

// ---------------- t_{k} = A * t_{k-1}, out += c_k * t_k ----------------
// 32-row tile per wave: K-tile and t-tile fragments shared across both halves.
// grid (NN/32=64 row tiles, 32 bh), block 32
__global__ void agf_attn_apply(const unsigned short* __restrict__ Qh,
                               const unsigned short* __restrict__ Kh,
                               const unsigned short* __restrict__ tin,
                               unsigned short* __restrict__ tout,
                               float* __restrict__ outacc,
                               const float* __restrict__ mbuf, const float* __restrict__ lbuf,
                               const float* __restrict__ coeffs, int kidx) {
    __shared__ unsigned short pbuf[32 * 32];   // P tile staged for C->A fragment reshape
    int lane = threadIdx.x & 31;
    int R0 = blockIdx.x * 32;
    int bh = blockIdx.y;
    int h = bh & 7;
    const bf16* qb = (const bf16*)(Qh + bh * NN * HD);
    const bf16* kb = (const bf16*)(Kh + bh * NN * HD);
    const bf16* tb = (const bf16*)(tin + bh * HD * NN);

    Frag16 aq[2][2];
#pragma unroll
    for (int hf = 0; hf < 2; ++hf) {
        int rowa = R0 + hf * 16 + (lane & 15);
        load_a16(aq[hf][0], qb,      rowa, HD, lane);
        load_a16(aq[hf][1], qb + 32, rowa, HD, lane);
    }

    float mrow[2][8], invl[2][8];
#pragma unroll
    for (int hf = 0; hf < 2; ++hf)
#pragma unroll
        for (int r = 0; r < 8; ++r) {
            int row = R0 + hf * 16 + r + ((lane >> 4) << 3);
            mrow[hf][r] = mbuf[bh * NN + row];
            invl[hf][r] = 1.0f / lbuf[bh * NN + row];
        }

    v8f acc[2][4];
#pragma unroll
    for (int hf = 0; hf < 2; ++hf)
#pragma unroll
        for (int j = 0; j < 4; ++j) acc[hf][j] = vzero8();

    for (int c0 = 0; c0 < NN; c0 += 32) {
        __syncthreads();
#pragma unroll
        for (int sub = 0; sub < 2; ++sub) {
            int key = c0 + sub * 16 + (lane & 15);
            Frag16 b0, b1;
            load_b16(b0, kb,      key, HD, lane);
            load_b16(b1, kb + 32, key, HD, lane);
#pragma unroll
            for (int hf = 0; hf < 2; ++hf) {
                v8f lg = vzero8();
                lg = wmma_bf(aq[hf][0], b0, lg);
                lg = wmma_bf(aq[hf][1], b1, lg);
#pragma unroll
                for (int r = 0; r < 8; ++r) {
                    float p = __expf(lg[r] - mrow[hf][r]) * invl[hf][r];
                    int trow = hf * 16 + r + ((lane >> 4) << 3);
                    pbuf[trow * 32 + sub * 16 + (lane & 15)] = f2bf(p);
                }
            }
        }
        __syncthreads();
        Frag16 ap[2];
#pragma unroll
        for (int hf = 0; hf < 2; ++hf)
            load_a16(ap[hf], (const bf16*)pbuf, hf * 16 + (lane & 15), 32, lane);
#pragma unroll
        for (int j = 0; j < 4; ++j) {
            int hd = j * 16 + (lane & 15);
            Frag16 bt;
            load_b16(bt, tb + c0, hd, NN, lane);   // t^T: [hd][n], k contiguous
#pragma unroll
            for (int hf = 0; hf < 2; ++hf)
                acc[hf][j] = wmma_bf(ap[hf], bt, acc[hf][j]);
        }
    }

    float ck = coeffs[h * 4 + kidx];
#pragma unroll
    for (int hf = 0; hf < 2; ++hf)
#pragma unroll
        for (int j = 0; j < 4; ++j) {
            int hd = j * 16 + (lane & 15);
#pragma unroll
            for (int r = 0; r < 8; ++r) {
                int token = R0 + hf * 16 + r + ((lane >> 4) << 3);
                int idx = (bh * HD + hd) * NN + token;
                float vl = acc[hf][j][r];
                tout[idx] = f2bf(vl);
                outacc[idx] += ck * vl;
            }
        }
}

// merged[b,n,h*64+hd] (bf16) <- outacc[b,h,hd,n] (f32)
__global__ void agf_merge(const float* __restrict__ outacc, unsigned short* __restrict__ merged) {
    int i = blockIdx.x * blockDim.x + threadIdx.x;   // MTOK*DIMM
    int token = i >> 9, c = i & 511;
    int b = token >> 11, n = token & 2047;
    int h = c >> 6, hd = c & 63;
    int bh = b * HH + h;
    merged[i] = f2bf(outacc[((bh * HD + hd) << 11) + n]);
}

// final projection: d_out = merged @ Wo + bo (fp32 row-major), 16x64 tile per wave
__global__ void agf_gemm_final(const unsigned short* __restrict__ merged,
                               const unsigned short* __restrict__ WoT,
                               const float* __restrict__ bo, float* __restrict__ out) {
    int lane = threadIdx.x & 31;
    int col0 = blockIdx.x * 64;
    int row0 = blockIdx.y * 16;
    const bf16* A  = (const bf16*)merged;
    const bf16* Bw = (const bf16*)WoT;
    int rowa = row0 + (lane & 15);
    v8f acc[4];
#pragma unroll
    for (int j = 0; j < 4; ++j) acc[j] = vzero8();

    for (int kk = 0; kk < DIMM; kk += 32) {
        Frag16 fa;
        load_a16(fa, A + kk, rowa, DIMM, lane);
#pragma unroll
        for (int j = 0; j < 4; ++j) {
            Frag16 fb;
            load_b16(fb, Bw + kk, col0 + j * 16 + (lane & 15), DIMM, lane);
            acc[j] = wmma_bf(fa, fb, acc[j]);
        }
    }
#pragma unroll
    for (int j = 0; j < 4; ++j) {
        int col = col0 + j * 16 + (lane & 15);
        float bval = bo[col];
#pragma unroll
        for (int r = 0; r < 8; ++r) {
            int token = row0 + r + ((lane >> 4) << 3);
            out[token * DIMM + col] = acc[j][r] + bval;
        }
    }
}

extern "C" void kernel_launch(void* const* d_in, const int* in_sizes, int n_in,
                              void* d_out, int out_size, void* d_ws, size_t ws_size,
                              hipStream_t stream) {
    const float* x  = (const float*)d_in[0];
    const float* Wq = (const float*)d_in[1];
    const float* bq = (const float*)d_in[2];
    const float* Wk = (const float*)d_in[3];
    const float* bk = (const float*)d_in[4];
    const float* Wv = (const float*)d_in[5];
    const float* bv = (const float*)d_in[6];
    const float* Wo = (const float*)d_in[7];
    const float* bo = (const float*)d_in[8];
    const float* coeffs = (const float*)d_in[9];
    float* outp = (float*)d_out;

    size_t off = 0;
    auto alloc = [&](size_t bytes) -> void* {
        void* p = (char*)d_ws + off;
        off += (bytes + 255) & ~(size_t)255;
        return p;
    };
    unsigned short* xb    = (unsigned short*)alloc((size_t)MTOK * DIMM * 2);     // 8 MB
    unsigned short* WTall = (unsigned short*)alloc((size_t)4 * DIMM * DIMM * 2); // 2 MB
    unsigned short* Qh    = (unsigned short*)alloc((size_t)BH * NN * HD * 2);    // 8 MB
    unsigned short* Kh    = (unsigned short*)alloc((size_t)BH * NN * HD * 2);    // 8 MB
    unsigned short* VT    = (unsigned short*)alloc((size_t)BH * NN * HD * 2);    // 8 MB
    float*          mbuf  = (float*)alloc((size_t)BH * NN * 4);                  // 256 KB
    float*          lbuf  = (float*)alloc((size_t)BH * NN * 4);                  // 256 KB
    unsigned short* tA    = (unsigned short*)alloc((size_t)BH * NN * HD * 2);    // 8 MB
    unsigned short* tB    = (unsigned short*)alloc((size_t)BH * NN * HD * 2);    // 8 MB
    float*          outacc= (float*)alloc((size_t)BH * HD * NN * 4);             // 16 MB
    unsigned short* merged= (unsigned short*)alloc((size_t)MTOK * DIMM * 2);     // 8 MB
    (void)ws_size; (void)n_in; (void)in_sizes; (void)out_size;

    const int NX = MTOK * DIMM;   // 4,194,304
    agf_cvt_x<<<NX / 256, 256, 0, stream>>>(x, xb, NX);
    agf_cvt_w<<<dim3((DIMM * DIMM) / 256, 4), 256, 0, stream>>>(Wq, Wk, Wv, Wo, WTall);

    agf_gemm_proj<<<dim3(DIMM / 64, MTOK / 16, 3), 32, 0, stream>>>(
        xb, WTall, bq, bk, bv, Qh, Kh, VT);

    agf_attn_stats<<<dim3(NN / 32, BH), 32, 0, stream>>>(Qh, Kh, mbuf, lbuf);

    agf_init_out<<<(BH * HD * NN) / 256, 256, 0, stream>>>(VT, outacc, coeffs);

    agf_attn_apply<<<dim3(NN / 32, BH), 32, 0, stream>>>(Qh, Kh, VT, tB, outacc, mbuf, lbuf, coeffs, 1);
    agf_attn_apply<<<dim3(NN / 32, BH), 32, 0, stream>>>(Qh, Kh, tB, tA, outacc, mbuf, lbuf, coeffs, 2);
    agf_attn_apply<<<dim3(NN / 32, BH), 32, 0, stream>>>(Qh, Kh, tA, tB, outacc, mbuf, lbuf, coeffs, 3);

    agf_merge<<<NX / 256, 256, 0, stream>>>(outacc, merged);

    agf_gemm_final<<<dim3(DIMM / 64, MTOK / 16), 32, 0, stream>>>(
        merged, WTall + 3 * DIMM * DIMM, bo, outp);
}
